// GraphStructureBuilder_58342835749143
// MI455X (gfx1250) — matrix operand order repaired and checked by reading, not compiled, not run on hardware
//
#include <hip/hip_runtime.h>
#include <hip/hip_bf16.h>

#define GN 8192
#define GD 128
#define THR 0.15f
#define CSPLIT 8           // column splits in pass 1

typedef __bf16 v16bf __attribute__((ext_vector_type(16)));
typedef __bf16 v8bf  __attribute__((ext_vector_type(8)));
typedef __bf16 v4bf  __attribute__((ext_vector_type(4)));
typedef float  v8f   __attribute__((ext_vector_type(8)));

// Load a 16-element bf16 fragment as two 16B chunks separated by `gap` elements.
__device__ __forceinline__ v16bf frag2(const __bf16* p, int gap) {
    union { v16bf v; v8bf h[2]; } u;
    u.h[0] = *(const v8bf*)(p);
    u.h[1] = *(const v8bf*)(p + gap);
    return u.v;
}

__device__ __forceinline__ v8f wmma_bf16(v16bf a, v16bf b, v8f c) {
    return __builtin_amdgcn_wmma_f32_16x16x32_bf16(
        /*neg_a=*/false, a, /*neg_b=*/false, b,
        /*c_mod=*/(short)0, c, /*reuse_a=*/false, /*reuse_b=*/false);
}

// Branchless sorted-top4 insert: max/min sorting chain (w0 >= w1 >= w2 >= w3).
__device__ __forceinline__ void top4_ins(float v, float& w0, float& w1,
                                         float& w2, float& w3) {
    float a = fmaxf(w0, v); float r = fminf(w0, v);
    float b = fmaxf(w1, r);       r = fminf(w1, r);
    float c = fmaxf(w2, r);       r = fminf(w2, r);
    float d = fmaxf(w3, r);
    w0 = a; w1 = b; w2 = c; w3 = d;
}

// ---------------------------------------------------------------------------
// Kernel 1: row L2-normalize, emit bf16 hi + residual lo (split precision).
// ---------------------------------------------------------------------------
__global__ __launch_bounds__(256)
void gsb_normalize(const float* __restrict__ x,
                   __bf16* __restrict__ hi, __bf16* __restrict__ lo) {
    const int lane = threadIdx.x & 31;
    const int row  = blockIdx.x * 8 + (threadIdx.x >> 5);
    const float4 v = ((const float4*)(x + (size_t)row * GD))[lane];
    float s = v.x * v.x + v.y * v.y + v.z * v.z + v.w * v.w;
    #pragma unroll
    for (int m = 16; m >= 1; m >>= 1) s += __shfl_xor(s, m, 32);
    const float inv = 1.0f / fmaxf(sqrtf(s), 1e-12f);

    float xs[4] = { v.x * inv, v.y * inv, v.z * inv, v.w * inv };
    v4bf h, l;
    #pragma unroll
    for (int i = 0; i < 4; ++i) {
        __bf16 hb = (__bf16)xs[i];
        h[i] = hb;
        l[i] = (__bf16)(xs[i] - (float)hb);
    }
    *(v4bf*)(hi + (size_t)row * GD + lane * 4) = h;
    *(v4bf*)(lo + (size_t)row * GD + lane * 4) = l;
}

// ---------------------------------------------------------------------------
// Kernel 2: per-row partial top-4 of off-diagonal sims over a column slice.
// One wave per (16-row strip, column slice). 3 bf16 WMMAs per K-chunk emulate
// fp32 (hi*hi + hi*lo + lo*hi), f32 accumulate. All 32 lanes scan: lane l
// owns row (l&15), half-wave selects columns 0-7 vs 8-15; merged via LDS.
// ---------------------------------------------------------------------------
__global__ __launch_bounds__(32)
void gsb_topk(const __bf16* __restrict__ hi, const __bf16* __restrict__ lo,
              float* __restrict__ part) {
    const int lane     = threadIdx.x;
    const int rowBase  = blockIdx.y * 16;
    const int colStart = blockIdx.x * (GN / CSPLIT);
    // Row stride 20 floats = 80 B: 16B-aligned rows (b128 LDS loads) and
    // conflict-free transpose (20*l mod 64 distinct for l in 0..15).
    __shared__ __align__(16) float lds[16][20];

    // A fragments for this 16-row strip (resident across the column loop).
    const int arow = rowBase + (lane & 15);
    const int ak   = (lane < 16) ? 0 : 8;
    const __bf16* ah = hi + (size_t)arow * GD;
    const __bf16* al = lo + (size_t)arow * GD;
    v16bf Ah[4], Al[4];
    #pragma unroll
    for (int kc = 0; kc < 4; ++kc) {
        Ah[kc] = frag2(ah + kc * 32 + ak, 16);
        Al[kc] = frag2(al + kc * 32 + ak, 16);
    }

    float w0 = -2.0f, w1 = -2.0f, w2 = -2.0f, w3 = -2.0f;
    const int mrow0 = (lane < 16) ? 0 : 8;
    const int ncol  = lane & 15;
    const int bk    = (lane < 16) ? 0 : 8;
    const int srow  = lane & 15;            // row this lane scans
    const int scol0 = (lane < 16) ? 0 : 8;  // column half this lane scans
    const int rowg  = rowBase + srow;       // global row for diagonal test

    for (int ct = 0; ct < (GN / CSPLIT) / 16; ++ct) {
        const int colBase = colStart + ct * 16;
        const int brow    = colBase + (lane & 15);
        const __bf16* bh = hi + (size_t)brow * GD;
        const __bf16* bl = lo + (size_t)brow * GD;

        v8f c = {0.f, 0.f, 0.f, 0.f, 0.f, 0.f, 0.f, 0.f};
        #pragma unroll
        for (int kc = 0; kc < 4; ++kc) {
            const int k0 = kc * 32 + bk;
            v16bf Bh = frag2(bh + k0, 16);
            v16bf Bl = frag2(bl + k0, 16);
            c = wmma_bf16(Ah[kc], Bh, c);
            c = wmma_bf16(Ah[kc], Bl, c);
            c = wmma_bf16(Al[kc], Bh, c);
        }

        __syncthreads();   // prior iteration's readers are done
        #pragma unroll
        for (int v = 0; v < 8; ++v) lds[mrow0 + v][ncol] = c[v];
        __syncthreads();

        // Branchless scan: 8 values per lane as two b128 LDS loads.
        const float4 q0 = *(const float4*)&lds[srow][scol0];
        const float4 q1 = *(const float4*)&lds[srow][scol0 + 4];
        const int cg = colBase + scol0;
        float e[8] = { q0.x, q0.y, q0.z, q0.w, q1.x, q1.y, q1.z, q1.w };
        #pragma unroll
        for (int n = 0; n < 8; ++n) {
            float s = (cg + n == rowg) ? -2.0f : e[n];  // mask diagonal
            top4_ins(s, w0, w1, w2, w3);
        }
    }

    // Merge the two half-wave top-4 lists per row through LDS.
    __syncthreads();
    if (lane >= 16) {
        float4 m = { w0, w1, w2, w3 };
        *(float4*)&lds[lane - 16][16] = m;
    }
    __syncthreads();
    if (lane < 16) {
        const float4 m = *(const float4*)&lds[lane][16];
        top4_ins(m.x, w0, w1, w2, w3);
        top4_ins(m.y, w0, w1, w2, w3);
        top4_ins(m.z, w0, w1, w2, w3);
        top4_ins(m.w, w0, w1, w2, w3);
        float4 o = { w0, w1, w2, w3 };
        *(float4*)(part + (((size_t)(rowBase + lane)) * CSPLIT + blockIdx.x) * 4) = o;
    }
}

// ---------------------------------------------------------------------------
// Kernel 3: merge the CSPLIT partial top-4 lists -> per-row 4th-largest value.
// ---------------------------------------------------------------------------
__global__ __launch_bounds__(256)
void gsb_merge(const float* __restrict__ part, float* __restrict__ t) {
    const int r = blockIdx.x * 256 + threadIdx.x;
    if (r >= GN) return;
    const float* p = part + (size_t)r * (CSPLIT * 4);
    float w0 = -2.0f, w1 = -2.0f, w2 = -2.0f, w3 = -2.0f;
    #pragma unroll
    for (int i = 0; i < CSPLIT; ++i) {
        const float4 q = ((const float4*)p)[i];
        top4_ins(q.x, w0, w1, w2, w3);
        top4_ins(q.y, w0, w1, w2, w3);
        top4_ins(q.z, w0, w1, w2, w3);
        top4_ins(q.w, w0, w1, w2, w3);
    }
    t[r] = w3;
}

// ---------------------------------------------------------------------------
// Kernel 4: recompute sim tiles (bit-identical math) and emit adj + weights
// with non-temporal stores (512 MB streamed once; keep hi/lo hot in L2).
// ---------------------------------------------------------------------------
__global__ __launch_bounds__(32)
void gsb_emit(const __bf16* __restrict__ hi, const __bf16* __restrict__ lo,
              const float* __restrict__ t, float* __restrict__ out) {
    const int lane     = threadIdx.x;
    const int rowBase  = blockIdx.y * 16;
    const int colChunk = blockIdx.x * 256;

    const int arow = rowBase + (lane & 15);
    const int ak   = (lane < 16) ? 0 : 8;
    const __bf16* ah = hi + (size_t)arow * GD;
    const __bf16* al = lo + (size_t)arow * GD;
    v16bf Ah[4], Al[4];
    #pragma unroll
    for (int kc = 0; kc < 4; ++kc) {
        Ah[kc] = frag2(ah + kc * 32 + ak, 16);
        Al[kc] = frag2(al + kc * 32 + ak, 16);
    }

    const int mrow0 = (lane < 16) ? 0 : 8;
    float thr[8];
    #pragma unroll
    for (int v = 0; v < 8; ++v) thr[v] = t[rowBase + mrow0 + v];

    float* __restrict__ adjO = out;
    float* __restrict__ wO   = out + (size_t)GN * GN;
    const int bk = (lane < 16) ? 0 : 8;

    for (int ct = 0; ct < 16; ++ct) {
        const int colBase = colChunk + ct * 16;
        const int brow    = colBase + (lane & 15);
        const __bf16* bh = hi + (size_t)brow * GD;
        const __bf16* bl = lo + (size_t)brow * GD;

        v8f c = {0.f, 0.f, 0.f, 0.f, 0.f, 0.f, 0.f, 0.f};
        #pragma unroll
        for (int kc = 0; kc < 4; ++kc) {
            const int k0 = kc * 32 + bk;
            v16bf Bh = frag2(bh + k0, 16);
            v16bf Bl = frag2(bl + k0, 16);
            c = wmma_bf16(Ah[kc], Bh, c);
            c = wmma_bf16(Ah[kc], Bl, c);
            c = wmma_bf16(Al[kc], Bh, c);
        }

        const int col = colBase + (lane & 15);
        #pragma unroll
        for (int v = 0; v < 8; ++v) {
            const int row  = rowBase + mrow0 + v;
            const float s  = c[v];
            const bool keep = (s > THR) && (s >= thr[v]) && (row != col);
            const size_t idx = (size_t)row * GN + col;
            __builtin_nontemporal_store(keep ? 1.0f : 0.0f, adjO + idx);
            __builtin_nontemporal_store(keep ? s    : 0.0f, wO + idx);
        }
    }
}

// ---------------------------------------------------------------------------
extern "C" void kernel_launch(void* const* d_in, const int* in_sizes, int n_in,
                              void* d_out, int out_size, void* d_ws, size_t ws_size,
                              hipStream_t stream) {
    (void)in_sizes; (void)n_in; (void)out_size; (void)ws_size;
    const float* x = (const float*)d_in[0];
    float* out = (float*)d_out;

    // Workspace carve-up (~5.3 MB total):
    //   hi: N*D bf16 (2 MB) | lo: N*D bf16 (2 MB) | t: N f32 | part: N*CSPLIT*4 f32 (1 MB)
    __bf16* hi  = (__bf16*)d_ws;
    __bf16* lo  = hi + (size_t)GN * GD;
    float*  t   = (float*)(lo + (size_t)GN * GD);
    float*  prt = t + GN;

    gsb_normalize<<<dim3(GN / 8), dim3(256), 0, stream>>>(x, hi, lo);
    gsb_topk<<<dim3(CSPLIT, GN / 16), dim3(32), 0, stream>>>(hi, lo, prt);
    gsb_merge<<<dim3(GN / 256), dim3(256), 0, stream>>>(prt, t);
    gsb_emit<<<dim3(GN / 256, GN / 16), dim3(32), 0, stream>>>(hi, lo, t, out);
}
